// SpatialGraph_57062935494848
// MI455X (gfx1250) — matrix operand order repaired and checked by reading, not compiled, not run on hardware
//
#include <hip/hip_runtime.h>
#include <hip/hip_bf16.h>
#include <math.h>

// ---------------------------------------------------------------------------
// CDNA5 (gfx1250) implementation: all GEMMs via v_wmma_f32_16x16x32_bf16,
// wave32 (warpSize==32), 8-wave blocks. Weights pre-transposed to bf16 [N][K]
// (L2-resident), B fragments loaded straight from global as b128s; A staged
// fp32->bf16 in LDS with vectorized float4 loads / b64 LDS stores.
// ---------------------------------------------------------------------------

typedef __attribute__((ext_vector_type(16))) __bf16 v16bf_t;
typedef __attribute__((ext_vector_type(8)))  __bf16 v8bf_t;
typedef __attribute__((ext_vector_type(4)))  __bf16 v4bf_t;
typedef __attribute__((ext_vector_type(8)))  float  v8f_t;

#define B_   16
#define TV_  128
#define NO_  10
#define NH_  4
#define T_   50
#define DT_  300
#define HID_ 512

#define GEMM_BM 64
#define GEMM_BN 64
#define GEMM_BK 32
#define LDS_STRIDE 40   // bf16 elements per LDS row (padded, even)

__device__ __forceinline__ v8f_t wmma_bf16(v16bf_t a, v16bf_t b, v8f_t c) {
  // 8 args: (neg_a, A, neg_b, B, c_mod, C, reuse_a, reuse_b)
  return __builtin_amdgcn_wmma_f32_16x16x32_bf16(false, a, false, b, (short)0, c, false, false);
}

__device__ __forceinline__ v16bf_t load_bfrag(const __bf16* p) {
  // 16 contiguous bf16 = 32B -> two global_load_b128
  v8bf_t lo = *(const v8bf_t*)p;
  v8bf_t hi = *(const v8bf_t*)(p + 8);
  v16bf_t r;
#pragma unroll
  for (int e = 0; e < 8; ++e) { r[e] = lo[e]; r[8 + e] = hi[e]; }
  return r;
}

__device__ __forceinline__ v16bf_t load_afrag(const __bf16* abase, int laneHalf) {
  // A fragment (16x32 bf16): lane = M row, VGPR v: K = half*16 + v*2
  v16bf_t afrag;
#pragma unroll
  for (int v = 0; v < 8; ++v) {
    int kidx = laneHalf * 16 + v * 2;
    afrag[2 * v]     = abase[kidx];
    afrag[2 * v + 1] = abase[kidx + 1];
  }
  return afrag;
}

// ---------------------------------------------------------------------------
// Generic fused GEMM:  C = act( ([C +] A(/arowDiv) @ W + biasVec + rowBias[b]) [* mulMat] )
//   A  : fp32 [M/arowDiv, K]
//   Wt : bf16 [N][Kpad] (transposed, zero-padded), row stride ldk
//   N must be a multiple of 64. act: 0 none, 1 tanh, 2 relu (mul before act).
// ---------------------------------------------------------------------------
__global__ __launch_bounds__(256)
void wmma_gemm_kernel(const float* __restrict__ A, int arowDiv, int lda,
                      const __bf16* __restrict__ Wt, int ldk,
                      float* __restrict__ C, int ldc,
                      int M, int N, int K, int Kpad,
                      const float* __restrict__ biasVec,
                      const float* __restrict__ rowBias, int rowsPerB,
                      const float* __restrict__ mulMat,
                      int accFlag, int act)
{
  __shared__ __bf16 As[GEMM_BM * LDS_STRIDE];
  const int tid      = threadIdx.x;
  const int lane     = tid & 31;
  const int wave     = tid >> 5;       // 0..7
  const int waveM    = wave & 3;       // 4 m-tiles of 16
  const int waveN    = wave >> 2;      // 2 n-halves of 32
  const int lane16   = lane & 15;
  const int laneHalf = lane >> 4;
  const int bm = blockIdx.x * GEMM_BM;
  const int bn = blockIdx.y * GEMM_BN;

  v8f_t acc0 = {};
  v8f_t acc1 = {};

  // Fast-path A row pointers (fixed across K loop): thread owns rows
  // row0 = tid>>3 and row1 = row0+32, float4 at kp = (tid&7)*4.
  const int  row0 = tid >> 3;
  const int  row1 = row0 + 32;
  const int  kp   = (tid & 7) * 4;
  int gr0 = bm + row0, gr1 = bm + row1;
  if (gr0 >= M) gr0 = M - 1;
  if (gr1 >= M) gr1 = M - 1;
  const long long aoff0 = (long long)((arowDiv == 1) ? gr0 : (gr0 / arowDiv)) * lda;
  const long long aoff1 = (long long)((arowDiv == 1) ? gr1 : (gr1 / arowDiv)) * lda;

  // B fragment base pointers (bf16, [N][Kpad]); N%64==0 so no n-guards.
  const __bf16* bptr0 = Wt + (long long)(bn + waveN * 32 + lane16)      * ldk + laneHalf * 16;
  const __bf16* bptr1 = Wt + (long long)(bn + waveN * 32 + 16 + lane16) * ldk + laneHalf * 16;

  const __bf16* abase = &As[(waveM * 16 + lane16) * LDS_STRIDE];

  // Block-uniform, loop-invariant path selection: interior M block and no K
  // tail -> branch-free steady-state loop (all big GEMMs; only the token GEMM
  // K=300 and M-tail blocks take the guarded loop).
  if (((bm + GEMM_BM) <= M) && (K % GEMM_BK == 0)) {
    for (int k0 = 0; k0 < K; k0 += GEMM_BK) {
      // Vectorized staging: 2 x float4 per thread -> pack4 bf16, ds_store_b64.
      const float4 f0 = *(const float4*)&A[aoff0 + k0 + kp];
      const float4 f1 = *(const float4*)&A[aoff1 + k0 + kp];
      v4bf_t p0, p1;
      p0[0] = (__bf16)f0.x; p0[1] = (__bf16)f0.y; p0[2] = (__bf16)f0.z; p0[3] = (__bf16)f0.w;
      p1[0] = (__bf16)f1.x; p1[1] = (__bf16)f1.y; p1[2] = (__bf16)f1.z; p1[3] = (__bf16)f1.w;
      *(v4bf_t*)&As[row0 * LDS_STRIDE + kp] = p0;
      *(v4bf_t*)&As[row1 * LDS_STRIDE + kp] = p1;
      __builtin_prefetch(&A[aoff0 + k0 + GEMM_BK + kp], 0, 1);  // global_prefetch_b8
      __syncthreads();

      v16bf_t afrag  = load_afrag(abase, laneHalf);
      v16bf_t bfrag0 = load_bfrag(bptr0 + k0);
      v16bf_t bfrag1 = load_bfrag(bptr1 + k0);
      acc0 = wmma_bf16(afrag, bfrag0, acc0);
      acc1 = wmma_bf16(afrag, bfrag1, acc1);
      __syncthreads();
    }
  } else {
    for (int k0 = 0; k0 < Kpad; k0 += GEMM_BK) {
      // Edge blocks / K tail: guarded scalar staging with zero fill.
#pragma unroll
      for (int i = 0; i < 8; ++i) {
        int idx = i * 256 + tid;
        int row = idx >> 5, kk = idx & 31;
        int gr = bm + row, gk = k0 + kk;
        float va = 0.0f;
        if (gr < M && gk < K) {
          int ar = (arowDiv == 1) ? gr : (gr / arowDiv);
          va = A[(long long)ar * lda + gk];
        }
        As[row * LDS_STRIDE + kk] = (__bf16)va;
      }
      __syncthreads();

      v16bf_t afrag  = load_afrag(abase, laneHalf);
      v16bf_t bfrag0 = load_bfrag(bptr0 + k0);
      v16bf_t bfrag1 = load_bfrag(bptr1 + k0);
      acc0 = wmma_bf16(afrag, bfrag0, acc0);
      acc1 = wmma_bf16(afrag, bfrag1, acc1);
      __syncthreads();
    }
  }

  // Epilogue: D layout — VGPR r holds M = r + 8*(lane/16), N = lane%16
#pragma unroll
  for (int r = 0; r < 8; ++r) {
    int gm = bm + waveM * 16 + laneHalf * 8 + r;
    if (gm >= M) continue;
    int gn0 = bn + waveN * 32 + lane16;
#pragma unroll
    for (int tle = 0; tle < 2; ++tle) {
      int gn = gn0 + tle * 16;
      if (gn >= N) continue;
      float v = (tle == 0) ? acc0[r] : acc1[r];
      long long off = (long long)gm * ldc + gn;
      if (accFlag) v += C[off];
      if (biasVec) v += biasVec[gn];
      if (rowBias) v += rowBias[(gm / rowsPerB) * N + gn];
      if (mulMat)  v *= mulMat[off];
      if (act == 1)      v = tanhf(v);
      else if (act == 2) v = fmaxf(v, 0.0f);
      C[off] = v;
    }
  }
}

// ---------------------------------------------------------------------------
// GRU scan: one 1024-thread block (32 waves). Wave w owns output columns
// j = 16w + lane%16 for all three gates (n-tiles {w, w+32, w+64}), so the
// r/z/n values for a column live in one lane and the recurrence state h[m][j]
// stays in registers. h is mirrored in LDS as bf16 for the WMMA A fragments.
// WhT: [1536][512] bf16, pre-transposed (L2-resident across 50 steps).
// ---------------------------------------------------------------------------
__global__ __launch_bounds__(1024)
void gru_scan_kernel(const float* __restrict__ xp,     // [16,50,1536] (has bx)
                     const __bf16* __restrict__ WhT,   // [1536,512] bf16
                     const float* __restrict__ bh,     // [1536]
                     float* __restrict__ gruOut)       // [16,50,512]
{
  __shared__ __bf16 hbf[16 * 520];
  const int tid      = threadIdx.x;
  const int lane     = tid & 31;
  const int wave     = tid >> 5;       // 0..31
  const int lane16   = lane & 15;
  const int laneHalf = lane >> 4;
  const int jcol     = wave * 16 + lane16;   // 0..511

  for (int i = tid; i < 16 * 520; i += 1024) hbf[i] = (__bf16)0.0f;
  float hold[8];
#pragma unroll
  for (int r = 0; r < 8; ++r) hold[r] = 0.0f;
  __syncthreads();

  const float bhj0 = bh[jcol];
  const float bhj1 = bh[jcol + 512];
  const float bhj2 = bh[jcol + 1024];

  const __bf16* wb0 = WhT + (long long)jcol          * 512 + laneHalf * 16;
  const __bf16* wb1 = WhT + (long long)(jcol + 512)  * 512 + laneHalf * 16;
  const __bf16* wb2 = WhT + (long long)(jcol + 1024) * 512 + laneHalf * 16;

  for (int t = 0; t < T_; ++t) {
    v8f_t acc0 = {}, acc1 = {}, acc2 = {};
    for (int kc = 0; kc < 16; ++kc) {
      int k0 = kc * 32;
      v16bf_t afrag = load_afrag(&hbf[lane16 * 520 + k0], laneHalf);
      v16bf_t bf0 = load_bfrag(wb0 + k0);
      v16bf_t bf1 = load_bfrag(wb1 + k0);
      v16bf_t bf2 = load_bfrag(wb2 + k0);
      acc0 = wmma_bf16(afrag, bf0, acc0);
      acc1 = wmma_bf16(afrag, bf1, acc1);
      acc2 = wmma_bf16(afrag, bf2, acc2);
    }
    __syncthreads();   // all hbf reads for step t done
#pragma unroll
    for (int r = 0; r < 8; ++r) {
      int m = laneHalf * 8 + r;    // batch index
      const float* xpr = &xp[((long long)m * T_ + t) * 1536 + jcol];
      float xr = xpr[0], xz = xpr[512], xn = xpr[1024];
      float hr = acc0[r] + bhj0;
      float hz = acc1[r] + bhj1;
      float hn = acc2[r] + bhj2;
      float rg = 1.0f / (1.0f + __expf(-(xr + hr)));
      float zg = 1.0f / (1.0f + __expf(-(xz + hz)));
      float ng = tanhf(xn + rg * hn);
      float hnew = (1.0f - zg) * ng + zg * hold[r];
      hold[r] = hnew;
      hbf[m * 520 + jcol] = (__bf16)hnew;
      gruOut[((long long)m * T_ + t) * 512 + jcol] = hnew;
    }
    __syncthreads();   // hbf writes visible before next step's fragments
  }
}

// --------------------------- small helper kernels ---------------------------

// in: fp32 [K,N] -> out: bf16 [N][Kpad], zero-padded K..Kpad
__global__ void conv_transpose_bf16(const float* __restrict__ in,
                                    __bf16* __restrict__ out,
                                    int K, int N, int Kpad)
{
  int idx = blockIdx.x * blockDim.x + threadIdx.x;
  if (idx < N * Kpad) {
    int n = idx / Kpad, k = idx % Kpad;
    float v = (k < K) ? in[(long long)k * N + n] : 0.0f;
    out[idx] = (__bf16)v;
  }
}

__global__ void pool_tanh_kernel(const float* __restrict__ gruOut,   // [16,50,512]
                                 const long long* __restrict__ lens,
                                 float* __restrict__ nodeQuery)      // [16,512]
{
  int idx = blockIdx.x * blockDim.x + threadIdx.x;
  if (idx < B_ * HID_) {
    int b = idx >> 9, j = idx & 511;
    int L = (int)lens[b];
    float s = 0.0f;
    for (int t = 0; t < L; ++t) s += gruOut[((long long)b * T_ + t) * 512 + j];
    nodeQuery[idx] = tanhf(s / (float)L);
  }
}

// Attention head (algebraically reduced): s_t = tokens[b,t,:].(qk @ q) + q.qkb
__global__ __launch_bounds__(512)
void attention_kernel(const float* __restrict__ nodeQuery,  // [16,512]
                      const float* __restrict__ gruOut,     // [16,50,512]
                      const float* __restrict__ tokens,     // [16,50,300]
                      const float* __restrict__ qhW, const float* __restrict__ qhB,  // [512,256],[256]
                      const float* __restrict__ qkW, const float* __restrict__ qkB,  // [300,256],[256]
                      const long long* __restrict__ lens,
                      float* __restrict__ attOut,           // [16,50]
                      float* __restrict__ nq)               // [16,512]
{
  __shared__ float qv[256];
  __shared__ float vv[304];
  __shared__ float cc;
  __shared__ float sarr[64];
  __shared__ float att[64];
  int b = blockIdx.x, tid = threadIdx.x;

  if (tid < 256) {
    float s = qhB[tid];
    for (int d = 0; d < 512; ++d) s += nodeQuery[b * 512 + d] * qhW[d * 256 + tid];
    qv[tid] = s;
  }
  __syncthreads();
  if (tid < 300) {
    float s = 0.0f;
    for (int n = 0; n < 256; ++n) s += qkW[tid * 256 + n] * qv[n];
    vv[tid] = s;
  } else if (tid == 300) {
    float s = 0.0f;
    for (int n = 0; n < 256; ++n) s += qkB[n] * qv[n];
    cc = s;
  }
  __syncthreads();
  int L = (int)lens[b];
  if (tid < T_) {
    float s = cc;
    const float* tk = &tokens[((long long)b * T_ + tid) * DT_];
    for (int d = 0; d < DT_; ++d) s += tk[d] * vv[d];
    sarr[tid] = s;
  }
  __syncthreads();
  if (tid == 0) {
    float mx = -3.0e38f;
    for (int t = 0; t < L; ++t) mx = fmaxf(mx, sarr[t]);
    float sum = 0.0f;
    for (int t = 0; t < T_; ++t) {
      float e = (t < L) ? __expf(sarr[t] - mx) : 0.0f;
      att[t] = e; sum += e;
    }
    for (int t = 0; t < T_; ++t) {
      float a = att[t] / sum + 1e-13f;
      att[t] = a;
      attOut[b * T_ + t] = a;
    }
  }
  __syncthreads();
  if (tid < 512) {
    float s = 0.0f;
    for (int t = 0; t < T_; ++t) s += att[t] * gruOut[((long long)b * T_ + t) * 512 + tid];
    nq[b * 512 + tid] = s;
  }
}

// Per-batch cat_lin bias table: out[b,n] = nq[b,:] @ W[0:512,:] + bvec[n]
__global__ void qw_kernel(const float* __restrict__ nq, const float* __restrict__ W,
                          const float* __restrict__ bvec, float* __restrict__ out)
{
  int idx = blockIdx.x * blockDim.x + threadIdx.x;
  if (idx < B_ * HID_) {
    int b = idx >> 9, n = idx & 511;
    float s = bvec[n];
    for (int d = 0; d < 512; ++d) s += nq[b * 512 + d] * W[d * 512 + n];
    out[idx] = s;
  }
}

// out[r,:] = sum_g in[r*G+g,:]
__global__ void groupsum_kernel(const float* __restrict__ in, float* __restrict__ out,
                                int rows, int G, int N)
{
  int idx = blockIdx.x * blockDim.x + threadIdx.x;
  if (idx < rows * N) {
    int r = idx / N, n = idx % N;
    float s = 0.0f;
    for (int g = 0; g < G; ++g) s += in[((long long)r * G + g) * N + n];
    out[idx] = s;
  }
}

// ------------------------------- host driver --------------------------------

static inline void launch_gemm(hipStream_t s, const float* A, int arowDiv, int lda,
                               const __bf16* Wt, int ldk, float* C, int ldc,
                               int M, int N, int K, int Kpad,
                               const float* biasVec, const float* rowBias, int rowsPerB,
                               const float* mulMat, int accFlag, int act)
{
  dim3 grid((M + GEMM_BM - 1) / GEMM_BM, (N + GEMM_BN - 1) / GEMM_BN);
  hipLaunchKernelGGL(wmma_gemm_kernel, grid, dim3(256), 0, s,
                     A, arowDiv, lda, Wt, ldk, C, ldc, M, N, K, Kpad,
                     biasVec, rowBias, rowsPerB, mulMat, accFlag, act);
}

extern "C" void kernel_launch(void* const* d_in, const int* in_sizes, int n_in,
                              void* d_out, int out_size, void* d_ws, size_t ws_size,
                              hipStream_t stream) {
  (void)in_sizes; (void)n_in; (void)out_size; (void)ws_size;

  const float* videoFeat = (const float*)d_in[0];
  const float* objects   = (const float*)d_in[2];
  const float* humans    = (const float*)d_in[4];
  const float* tokens    = (const float*)d_in[6];
  const long long* tokLen = (const long long*)d_in[7];
  const float* gru_Wx = (const float*)d_in[8];
  const float* gru_Wh = (const float*)d_in[9];
  const float* gru_bx = (const float*)d_in[10];
  const float* gru_bh = (const float*)d_in[11];
  const float* fv_W = (const float*)d_in[12];  const float* fv_b = (const float*)d_in[13];
  const float* fo_W = (const float*)d_in[14];  const float* fo_b = (const float*)d_in[15];
  const float* fh_W = (const float*)d_in[16];  const float* fh_b = (const float*)d_in[17];
  const float* vq_W = (const float*)d_in[18];  const float* vq_b = (const float*)d_in[19];
  const float* hq_W = (const float*)d_in[20];  const float* hq_b = (const float*)d_in[21];
  const float* oq_W = (const float*)d_in[22];  const float* oq_b = (const float*)d_in[23];
  const float* fvhq_W = (const float*)d_in[24]; const float* fvhq_b = (const float*)d_in[25];
  const float* fvoq_W = (const float*)d_in[26]; const float* fvoq_b = (const float*)d_in[27];
  const float* fuv_W = (const float*)d_in[28];  const float* fuv_b = (const float*)d_in[29];
  const float* fohq_W = (const float*)d_in[30]; const float* fohq_b = (const float*)d_in[31];
  const float* fuo_W = (const float*)d_in[32];  const float* fuo_b = (const float*)d_in[33];
  const float* fuh_W = (const float*)d_in[34];  const float* fuh_b = (const float*)d_in[35];
  const float* qh1_W = (const float*)d_in[36];  const float* qh1_b = (const float*)d_in[37];
  const float* qh2_W = (const float*)d_in[38];  const float* qh2_b = (const float*)d_in[39];
  const float* qh3_W = (const float*)d_in[40];  const float* qh3_b = (const float*)d_in[41];
  const float* qk1_W = (const float*)d_in[42];  const float* qk1_b = (const float*)d_in[43];
  const float* qk2_W = (const float*)d_in[44];  const float* qk2_b = (const float*)d_in[45];
  const float* qk3_W = (const float*)d_in[46];  const float* qk3_b = (const float*)d_in[47];

  const int MV = B_ * TV_;            // 2048
  const int MO = B_ * TV_ * NO_;      // 20480
  const int MH = B_ * TV_ * NH_;      // 8192

  float* out = (float*)d_out;
  float* uV = out;                                   // [16,128,512]
  float* uO = out + (long long)MV * HID_;            // [16,128,10,512]
  float* uH = uO + (long long)MO * HID_;             // [16,128,4,512]
  float* attHO = uH + (long long)MH * HID_;          // [16,50]
  float* attVH = attHO + B_ * T_;
  float* attVO = attVH + B_ * T_;

  char* wsb = (char*)d_ws;
  size_t off = 0;
  auto alloc = [&](size_t bytes) -> void* {
    void* p = (void*)(wsb + off);
    off += (bytes + 255) & ~(size_t)255;
    return p;
  };
  float*  xp        = (float*)alloc((size_t)B_ * T_ * 1536 * 4);
  float*  gruOut    = (float*)alloc((size_t)B_ * T_ * HID_ * 4);
  float*  nodeQuery = (float*)alloc((size_t)B_ * HID_ * 4);
  float*  nqHO      = (float*)alloc((size_t)B_ * HID_ * 4);
  float*  nqVO      = (float*)alloc((size_t)B_ * HID_ * 4);
  float*  nqVH      = (float*)alloc((size_t)B_ * HID_ * 4);
  float*  qwVHvq    = (float*)alloc((size_t)B_ * HID_ * 4);
  float*  qwVHhq    = (float*)alloc((size_t)B_ * HID_ * 4);
  float*  qwVOvq    = (float*)alloc((size_t)B_ * HID_ * 4);
  float*  qwVOoq    = (float*)alloc((size_t)B_ * HID_ * 4);
  float*  qwHOoq    = (float*)alloc((size_t)B_ * HID_ * 4);
  float*  qwHOhq    = (float*)alloc((size_t)B_ * HID_ * 4);
  float*  nodeVideo  = (float*)alloc((size_t)MV * HID_ * 4);
  float*  nodeObject = (float*)alloc((size_t)MO * HID_ * 4);
  float*  nodeHuman  = (float*)alloc((size_t)MH * HID_ * 4);
  float*  bufVQ  = (float*)alloc((size_t)MV * HID_ * 4);
  float*  bufHQ  = (float*)alloc((size_t)MH * HID_ * 4);
  float*  bufOQ  = (float*)alloc((size_t)MO * HID_ * 4);
  float*  bufSum = (float*)alloc((size_t)MV * HID_ * 4);
  float*  bufT1  = (float*)alloc((size_t)MO * HID_ * 4);
  float*  bufT2  = (float*)alloc((size_t)MO * HID_ * 4);
  // bf16 transposed weights [N][Kpad]
  __bf16* WxT   = (__bf16*)alloc((size_t)1536 * 320 * 2);
  __bf16* WhT   = (__bf16*)alloc((size_t)1536 * 512 * 2);
  __bf16* fvT   = (__bf16*)alloc((size_t)512 * 1024 * 2);
  __bf16* fhT   = (__bf16*)alloc((size_t)512 * 2048 * 2);
  __bf16* foT   = (__bf16*)alloc((size_t)512 * 2048 * 2);
  __bf16* vqT   = (__bf16*)alloc((size_t)512 * 1024 * 2);
  __bf16* hqT   = (__bf16*)alloc((size_t)512 * 1024 * 2);
  __bf16* oqT   = (__bf16*)alloc((size_t)512 * 1024 * 2);
  __bf16* fvhqT = (__bf16*)alloc((size_t)512 * 1024 * 2);
  __bf16* fvoqT = (__bf16*)alloc((size_t)512 * 1024 * 2);
  __bf16* fohqT = (__bf16*)alloc((size_t)512 * 1024 * 2);
  __bf16* fuvT  = (__bf16*)alloc((size_t)512 * 512 * 2);
  __bf16* fuoT  = (__bf16*)alloc((size_t)512 * 512 * 2);
  __bf16* fuhT  = (__bf16*)alloc((size_t)512 * 512 * 2);

  auto launch_tr = [&](const float* in, __bf16* o, int K, int N, int Kpad) {
    int tot = N * Kpad;
    hipLaunchKernelGGL(conv_transpose_bf16, dim3((tot + 255) / 256), dim3(256), 0, stream,
                       in, o, K, N, Kpad);
  };
  launch_tr(gru_Wx, WxT, 300, 1536, 320);
  launch_tr(gru_Wh, WhT, 512, 1536, 512);
  launch_tr(fv_W,  fvT,  1024, 512, 1024);
  launch_tr(fh_W,  fhT,  2048, 512, 2048);
  launch_tr(fo_W,  foT,  2048, 512, 2048);
  launch_tr(vq_W,  vqT,  1024, 512, 1024);
  launch_tr(hq_W,  hqT,  1024, 512, 1024);
  launch_tr(oq_W,  oqT,  1024, 512, 1024);
  launch_tr(fvhq_W, fvhqT, 1024, 512, 1024);
  launch_tr(fvoq_W, fvoqT, 1024, 512, 1024);
  launch_tr(fohq_W, fohqT, 1024, 512, 1024);
  launch_tr(fuv_W, fuvT, 512, 512, 512);
  launch_tr(fuo_W, fuoT, 512, 512, 512);
  launch_tr(fuh_W, fuhT, 512, 512, 512);

  // Transposed halves of the cat_lin weights: rows 0..511 -> k offset 0,
  // rows 512..1023 -> k offset 512 (row stride 1024).
  const __bf16* vqTb   = vqT + 512;
  const __bf16* hqTb   = hqT + 512;
  const __bf16* oqTb   = oqT + 512;
  const __bf16* fvhqTb = fvhqT + 512;
  const __bf16* fvoqTb = fvoqT + 512;
  const __bf16* fohqTb = fohqT + 512;

  // --- GRU ---
  launch_gemm(stream, tokens, 1, DT_, WxT, 320, xp, 1536,
              B_ * T_, 1536, DT_, 320, gru_bx, nullptr, 1, nullptr, 0, 0);
  hipLaunchKernelGGL(gru_scan_kernel, dim3(1), dim3(1024), 0, stream, xp, WhT, gru_bh, gruOut);
  hipLaunchKernelGGL(pool_tanh_kernel, dim3((B_ * HID_ + 255) / 256), dim3(256), 0, stream,
                     gruOut, tokLen, nodeQuery);

  // --- attention heads (HO, VO, VH) ---
  hipLaunchKernelGGL(attention_kernel, dim3(B_), dim3(512), 0, stream,
                     nodeQuery, gruOut, tokens, qh1_W, qh1_b, qk1_W, qk1_b, tokLen, attHO, nqHO);
  hipLaunchKernelGGL(attention_kernel, dim3(B_), dim3(512), 0, stream,
                     nodeQuery, gruOut, tokens, qh2_W, qh2_b, qk2_W, qk2_b, tokLen, attVO, nqVO);
  hipLaunchKernelGGL(attention_kernel, dim3(B_), dim3(512), 0, stream,
                     nodeQuery, gruOut, tokens, qh3_W, qh3_b, qk3_W, qk3_b, tokLen, attVH, nqVH);

  // --- node embeddings (tanh GEMMs) ---
  launch_gemm(stream, videoFeat, 1, 1024, fvT, 1024, nodeVideo, 512,
              MV, 512, 1024, 1024, fv_b, nullptr, 1, nullptr, 0, 1);
  launch_gemm(stream, objects, 1, 2048, fhT, 2048, nodeObject, 512,   // factor_human on objects
              MO, 512, 2048, 2048, fh_b, nullptr, 1, nullptr, 0, 1);
  launch_gemm(stream, humans, 1, 2048, foT, 2048, nodeHuman, 512,     // factor_obj on humans
              MH, 512, 2048, 2048, fo_b, nullptr, 1, nullptr, 0, 1);

  // --- per-batch cat_lin bias tables: nq @ W_top + b ---
  dim3 qg((B_ * HID_ + 255) / 256), qb(256);
  hipLaunchKernelGGL(qw_kernel, qg, qb, 0, stream, nqVH, vq_W, vq_b, qwVHvq);
  hipLaunchKernelGGL(qw_kernel, qg, qb, 0, stream, nqVH, hq_W, hq_b, qwVHhq);
  hipLaunchKernelGGL(qw_kernel, qg, qb, 0, stream, nqVO, vq_W, vq_b, qwVOvq);
  hipLaunchKernelGGL(qw_kernel, qg, qb, 0, stream, nqVO, oq_W, oq_b, qwVOoq);
  hipLaunchKernelGGL(qw_kernel, qg, qb, 0, stream, nqHO, oq_W, oq_b, qwHOoq);
  hipLaunchKernelGGL(qw_kernel, qg, qb, 0, stream, nqHO, hq_W, hq_b, qwHOhq);

  // --- initialize uV/uO/uH = node embeddings ---
  hipMemcpyAsync(uV, nodeVideo,  (size_t)MV * HID_ * 4, hipMemcpyDeviceToDevice, stream);
  hipMemcpyAsync(uO, nodeObject, (size_t)MO * HID_ * 4, hipMemcpyDeviceToDevice, stream);
  hipMemcpyAsync(uH, nodeHuman,  (size_t)MH * HID_ * 4, hipMemcpyDeviceToDevice, stream);

  dim3 gsg((MV * HID_ + 255) / 256), gsb(256);

  for (int it = 0; it < 2; ++it) {
    // ---------------- video update ----------------
    launch_gemm(stream, uV, 1, 512, vqTb, 1024, bufVQ, 512, MV, 512, 512, 512,
                nullptr, qwVHvq, TV_, nullptr, 0, 0);
    launch_gemm(stream, uH, 1, 512, hqTb, 1024, bufHQ, 512, MH, 512, 512, 512,
                nullptr, qwVHhq, TV_ * NH_, nullptr, 0, 0);
    hipLaunchKernelGGL(groupsum_kernel, gsg, gsb, 0, stream, bufHQ, bufSum, MV, NH_, 512);
    launch_gemm(stream, bufVQ, 1, 512, fvhqT, 1024, bufT1, 512, MV, 512, 512, 512,
                nullptr, nullptr, 1, nullptr, 0, 0);
    launch_gemm(stream, bufSum, 1, 512, fvhqTb, 1024, bufT1, 512, MV, 512, 512, 512,
                fvhq_b, nullptr, 1, nullptr, 1, 0);
    launch_gemm(stream, uV, 1, 512, vqTb, 1024, bufVQ, 512, MV, 512, 512, 512,
                nullptr, qwVOvq, TV_, nullptr, 0, 0);
    launch_gemm(stream, uO, 1, 512, oqTb, 1024, bufOQ, 512, MO, 512, 512, 512,
                nullptr, qwVOoq, TV_ * NO_, nullptr, 0, 0);
    hipLaunchKernelGGL(groupsum_kernel, gsg, gsb, 0, stream, bufOQ, bufSum, MV, NO_, 512);
    launch_gemm(stream, bufVQ, 1, 512, fvoqT, 1024, bufT2, 512, MV, 512, 512, 512,
                nullptr, nullptr, 1, nullptr, 0, 0);
    launch_gemm(stream, bufSum, 1, 512, fvoqTb, 1024, bufT2, 512, MV, 512, 512, 512,
                fvoq_b, nullptr, 1, bufT1, 1, 0);
    launch_gemm(stream, bufT2, 1, 512, fuvT, 512, uV, 512, MV, 512, 512, 512,
                fuv_b, nullptr, 1, nodeVideo, 0, 2);

    // ---------------- object update ----------------
    launch_gemm(stream, uO, 1, 512, oqTb, 1024, bufOQ, 512, MO, 512, 512, 512,
                nullptr, qwHOoq, TV_ * NO_, nullptr, 0, 0);
    launch_gemm(stream, uH, 1, 512, hqTb, 1024, bufHQ, 512, MH, 512, 512, 512,
                nullptr, qwHOhq, TV_ * NH_, nullptr, 0, 0);
    hipLaunchKernelGGL(groupsum_kernel, gsg, gsb, 0, stream, bufHQ, bufSum, MV, NH_, 512);
    launch_gemm(stream, bufOQ, 1, 512, fohqT, 1024, bufT1, 512, MO, 512, 512, 512,
                nullptr, nullptr, 1, nullptr, 0, 0);
    launch_gemm(stream, bufSum, NO_, 512, fohqTb, 1024, bufT1, 512, MO, 512, 512, 512,
                fohq_b, nullptr, 1, nullptr, 1, 0);
    launch_gemm(stream, uO, 1, 512, oqTb, 1024, bufOQ, 512, MO, 512, 512, 512,
                nullptr, qwVOoq, TV_ * NO_, nullptr, 0, 0);
    launch_gemm(stream, uV, 1, 512, vqTb, 1024, bufVQ, 512, MV, 512, 512, 512,
                nullptr, qwVOvq, TV_, nullptr, 0, 0);
    launch_gemm(stream, bufOQ, 1, 512, fvoqT, 1024, bufT2, 512, MO, 512, 512, 512,
                nullptr, nullptr, 1, nullptr, 0, 0);
    launch_gemm(stream, bufVQ, NO_, 512, fvoqTb, 1024, bufT2, 512, MO, 512, 512, 512,
                fvoq_b, nullptr, 1, bufT1, 1, 0);
    launch_gemm(stream, bufT2, 1, 512, fuoT, 512, uO, 512, MO, 512, 512, 512,
                fuo_b, nullptr, 1, nodeObject, 0, 2);

    // ---------------- human update ----------------
    launch_gemm(stream, uO, 1, 512, oqTb, 1024, bufOQ, 512, MO, 512, 512, 512,
                nullptr, qwHOoq, TV_ * NO_, nullptr, 0, 0);
    launch_gemm(stream, uH, 1, 512, hqTb, 1024, bufHQ, 512, MH, 512, 512, 512,
                nullptr, qwHOhq, TV_ * NH_, nullptr, 0, 0);
    hipLaunchKernelGGL(groupsum_kernel, gsg, gsb, 0, stream, bufOQ, bufSum, MV, NO_, 512);
    launch_gemm(stream, bufHQ, 1, 512, fohqT, 1024, bufT1, 512, MH, 512, 512, 512,
                nullptr, nullptr, 1, nullptr, 0, 0);
    launch_gemm(stream, bufSum, NH_, 512, fohqTb, 1024, bufT1, 512, MH, 512, 512, 512,
                fohq_b, nullptr, 1, nullptr, 1, 0);
    launch_gemm(stream, uV, 1, 512, vqTb, 1024, bufVQ, 512, MV, 512, 512, 512,
                nullptr, qwVHvq, TV_, nullptr, 0, 0);
    launch_gemm(stream, uH, 1, 512, hqTb, 1024, bufHQ, 512, MH, 512, 512, 512,
                nullptr, qwVHhq, TV_ * NH_, nullptr, 0, 0);
    launch_gemm(stream, bufHQ, 1, 512, fvhqT, 1024, bufT2, 512, MH, 512, 512, 512,
                nullptr, nullptr, 1, nullptr, 0, 0);
    launch_gemm(stream, bufVQ, NH_, 512, fvhqTb, 1024, bufT2, 512, MH, 512, 512, 512,
                fvhq_b, nullptr, 1, bufT1, 1, 0);
    launch_gemm(stream, bufT2, 1, 512, fuhT, 512, uH, 512, MH, 512, 512, 512,
                fuh_b, nullptr, 1, nodeHuman, 0, 2);
  }
}
